// MultiHeadGraphAttention_18004502905170
// MI455X (gfx1250) — matrix operand order repaired and checked by reading, not compiled, not run on hardware
//
#include <hip/hip_runtime.h>
#include <hip/hip_bf16.h>

// MultiHeadGraphAttention for MI455X (gfx1250), wave32 + v_wmma_f32_16x16x32_bf16
// + Tensor Data Mover (tensor_load_to_lds) staging in the hot edge-attention loop.
// B=2, T=128, H=8, D=64, DE=16, C=512, E=128.

#define BB 2
#define TT 128
#define HH 8
#define DD 64
#define DE 16
#define CC 512
#define EE 128

typedef __attribute__((ext_vector_type(16))) __bf16 v16bf;
typedef __attribute__((ext_vector_type(8)))  float  v8f;

union BF16Vec { v16bf v; unsigned short u[16]; uint4 q[2]; };

__device__ __forceinline__ unsigned short f2bf(float f) {
  unsigned int x = __float_as_uint(f);
  x += 0x7FFFu + ((x >> 16) & 1u);          // round-to-nearest-even
  return (unsigned short)(x >> 16);
}

// A operand: lane holds row (lane&15); element i covers K = i + 8*half + (i>=8 ? 8 : 0)
__device__ __forceinline__ int a_kidx(int i, int half) { return i + 8 * half + (i >= 8 ? 8 : 0); }
// B operand: lane holds col (lane&15); element i covers K = i + 8*half + (i>=8 ? 8 : 0)
__device__ __forceinline__ int b_kidx(int i, int half) { return i + 8 * half + (i >= 8 ? 8 : 0); }

__device__ __forceinline__ v8f wmma_bf16(v16bf a, v16bf b, v8f c) {
  return __builtin_amdgcn_wmma_f32_16x16x32_bf16(false, a, false, b, (short)0, c, false, false);
}

// convert 8 floats (two float4) into bf16 slots [at..at+7]
__device__ __forceinline__ void cvt8(BF16Vec& dst, int at, const float4 a, const float4 b) {
  dst.u[at + 0] = f2bf(a.x); dst.u[at + 1] = f2bf(a.y);
  dst.u[at + 2] = f2bf(a.z); dst.u[at + 3] = f2bf(a.w);
  dst.u[at + 4] = f2bf(b.x); dst.u[at + 5] = f2bf(b.y);
  dst.u[at + 6] = f2bf(b.z); dst.u[at + 7] = f2bf(b.w);
}

// A tile from f32 row-major, contiguous K chunk (ap = &A[row][k0]), 16B aligned.
__device__ __forceinline__ void load_a_f32(BF16Vec& av, const float* ap, int half) {
  const float4* p0 = (const float4*)(ap + 8 * half);
  const float4* p1 = (const float4*)(ap + 16 + 8 * half);
  cvt8(av, 0, p0[0], p0[1]);
  cvt8(av, 8, p1[0], p1[1]);
}

// A tile from bf16 row-major, contiguous K chunk (ap = &A[row][k0]), 16B aligned.
__device__ __forceinline__ void load_a_bf(BF16Vec& av, const unsigned short* ap, int half) {
  av.q[0] = *(const uint4*)(ap + 8 * half);
  av.q[1] = *(const uint4*)(ap + 16 + 8 * half);
}

// ---------------------------------------------------------------------------
// Tensor Data Mover: stage a 2D bf16 tile (128 rows x 16 elems, row stride rs)
// from global into LDS.  D# per CDNA5 ISA ch.8 (async_tensor).
// This toolchain's builtin takes 6 args:
//   (uint32x4 g0, int32x8 g1, int32x4 g2, int32x4 g3, int32x8 extra, i32 cpol)
// ---------------------------------------------------------------------------
#if __has_builtin(__builtin_amdgcn_tensor_load_to_lds) && __has_builtin(__builtin_amdgcn_s_wait_tensorcnt)
#define USE_TDM 1
typedef __attribute__((ext_vector_type(4))) unsigned int v4u;
typedef __attribute__((ext_vector_type(8))) int v8i;
typedef __attribute__((ext_vector_type(4))) int v4i;

__device__ __forceinline__ void tdm_load_tile(unsigned short* ldsDst,
                                              const unsigned short* gsrc,
                                              int row_stride) {
  unsigned long long ga = (unsigned long long)(size_t)gsrc;
  unsigned int la = (unsigned int)(size_t)ldsDst;   // low 32 bits = LDS offset
  v4u g0;
  g0[0] = 1u;                                       // count=1 valid descriptor
  g0[1] = la;                                       // lds_addr
  g0[2] = (unsigned int)ga;                         // global_addr[31:0]
  g0[3] = (unsigned int)(ga >> 32) | (2u << 30);    // global_addr[56:32] | type=2
  v8i g1;
  g1[0] = (int)(1u << 16);                          // wg_mask=0, data_size=1 (2B)
  g1[1] = (int)((unsigned)DE << 16);                // tensor_dim0 = 16
  g1[2] = (int)((unsigned)TT << 16);                // tensor_dim1 = 128
  g1[3] = (int)((unsigned)DE << 16);                // tile_dim0 = 16
  g1[4] = TT;                                       // tile_dim1 = 128, tile_dim2 = 0
  g1[5] = row_stride;                               // tensor_dim0_stride[31:0]
  g1[6] = 0;                                        // stride hi / dim1_stride lo
  g1[7] = 0;
  v4i z = {0, 0, 0, 0};
  v8i z8 = {0, 0, 0, 0, 0, 0, 0, 0};
  __builtin_amdgcn_tensor_load_to_lds(g0, g1, z, z, z8, 0);
}
#else
#define USE_TDM 0
#endif

// ---------------------------------------------------------------------------
// K1: q,k,v = x1 @ W* + b*  -> bf16 (B,H,T,D).  M=256, K=64, N=512, 3 mats.
// ---------------------------------------------------------------------------
__global__ void __launch_bounds__(128) qkv_kernel(
    const float* __restrict__ x1,
    const float* __restrict__ Wq, const float* __restrict__ bq,
    const float* __restrict__ Wk, const float* __restrict__ bk,
    const float* __restrict__ Wv, const float* __restrict__ bv,
    unsigned short* __restrict__ qb, unsigned short* __restrict__ kb,
    unsigned short* __restrict__ vb)
{
  int gw = blockIdx.x * 4 + (threadIdx.x >> 5);   // 0..1535
  int lane = threadIdx.x & 31, half = lane >> 4, lr = lane & 15;
  int mat = gw >> 9;
  int rem = gw & 511;
  int mt = rem >> 5, nt = rem & 31;
  const float* W  = (mat == 0) ? Wq : (mat == 1) ? Wk : Wv;
  const float* bi = (mat == 0) ? bq : (mat == 1) ? bk : bv;
  unsigned short* dst = (mat == 0) ? qb : (mat == 1) ? kb : vb;
  int m0 = mt * 16;
  v8f acc = {};
  #pragma unroll
  for (int kc = 0; kc < 2; ++kc) {
    BF16Vec av, bvv;
    load_a_f32(av, x1 + (m0 + lr) * DD + kc * 32, half);
    #pragma unroll
    for (int i = 0; i < 16; ++i)
      bvv.u[i] = f2bf(W[(kc * 32 + b_kidx(i, half)) * CC + nt * 16 + lr]);
    acc = wmma_bf16(av.v, bvv.v, acc);
  }
  int n = nt * 16 + lr, h = n >> 6, d = n & 63;
  float bn = bi[n];
  #pragma unroll
  for (int r = 0; r < 8; ++r) {
    int m = m0 + r + 8 * half;
    int b = m >> 7, t = m & 127;
    dst[(((b * HH + h) * TT) + t) * DD + d] = f2bf(acc[r] + bn);
  }
}

// ---------------------------------------------------------------------------
// K2: scaled_qk[bh,s,t] = q[bh,s,:] . k[bh,t,:] / 8   (128x128x64 per bh)
// ---------------------------------------------------------------------------
__global__ void __launch_bounds__(128) sqk_kernel(
    const unsigned short* __restrict__ qb, const unsigned short* __restrict__ kb,
    float* __restrict__ sqk)
{
  int gw = blockIdx.x * 4 + (threadIdx.x >> 5);   // 0..1023
  int lane = threadIdx.x & 31, half = lane >> 4, lr = lane & 15;
  int nt = gw & 7, mt = (gw >> 3) & 7, bh = gw >> 6;
  const unsigned short* Q  = qb + bh * TT * DD;
  const unsigned short* Kt = kb + bh * TT * DD;
  int m0 = mt * 16;
  v8f acc = {};
  #pragma unroll
  for (int kc = 0; kc < 2; ++kc) {
    BF16Vec av, bvv;
    load_a_bf(av, Q + (m0 + lr) * DD + kc * 32, half);
    const unsigned short* bp = Kt + (nt * 16 + lr) * DD + kc * 32 + 8 * half;
    bvv.q[0] = *(const uint4*)(bp);               // B[k][n] = k[n][k], K = 8*half..+7, 16+8*half..+7
    bvv.q[1] = *(const uint4*)(bp + 16);
    acc = wmma_bf16(av.v, bvv.v, acc);
  }
  int n = nt * 16 + lr;
  #pragma unroll
  for (int r = 0; r < 8; ++r) {
    int m = m0 + r + 8 * half;
    sqk[bh * TT * TT + m * TT + n] = acc[r] * 0.125f;
  }
}

// ---------------------------------------------------------------------------
// K3: eq/ev = e @ Weq/Wev + b (K=16); ek = scaled_qk(H-gather) @ Wek + bek (K=8)
//     all written bf16 in (B,H,T1,T2,DE).
// ---------------------------------------------------------------------------
__global__ void __launch_bounds__(256) edge_proj_kernel(
    const float* __restrict__ e, const float* __restrict__ sqk,
    const float* __restrict__ Weq, const float* __restrict__ beq,
    const float* __restrict__ Wev, const float* __restrict__ bev,
    const float* __restrict__ Wek, const float* __restrict__ bek,
    unsigned short* __restrict__ eq, unsigned short* __restrict__ ev,
    unsigned short* __restrict__ ek)
{
  int idx = blockIdx.x * 256 + threadIdx.x;       // B*T*T*128
  int n = idx & 127;
  int r = idx >> 7;                               // (b,t2,t1)
  int t1 = r & 127, t2 = (r >> 7) & 127, b = r >> 14;
  const float* ein = e + r * DE;
  float aq = beq[n], avv = bev[n];
  #pragma unroll
  for (int c = 0; c < 16; ++c) {
    float x = ein[c];
    aq  += x * Weq[c * EE + n];
    avv += x * Wev[c * EE + n];
  }
  float ak = bek[n];
  #pragma unroll
  for (int hh = 0; hh < 8; ++hh) {
    float x = sqk[((b * HH + hh) * TT + t2) * TT + t1];
    ak += x * Wek[hh * EE + n];
  }
  int h = n >> 4, de = n & 15;
  int o = (((b * HH + h) * TT + t1) * TT + t2) * DE + de;
  eq[o] = f2bf(aq); ev[o] = f2bf(avv); ek[o] = f2bf(ak);
}

// ---------------------------------------------------------------------------
// K4: edge attention. One block (256 thr, 8 waves) per (b,h,t).
// mode 0: rows over t2 (stride DE)      -> qk1/out1, writes eo
// mode 1: rows over t1 (stride T*DE)    -> qk2/out2, accumulates eo
// eq/ek/ev tiles (128x16 bf16 each) staged into LDS via TDM (tensor_load_to_lds,
// TENSORcnt) so 8 waves reuse them.  Per wave: 16x128 logits = 8 WMMA (K=16
// padded to 32), shfl softmax, P restaged via LDS, out = P @ ev = 4 WMMA.
// ---------------------------------------------------------------------------
__global__ void __launch_bounds__(256) edge_attn_kernel(
    const unsigned short* __restrict__ eq, const unsigned short* __restrict__ ek,
    const unsigned short* __restrict__ ev, float* __restrict__ eo, int mode)
{
  __shared__ __align__(16) unsigned short eqL[TT * DE];
  __shared__ __align__(16) unsigned short ekL[TT * DE];
  __shared__ __align__(16) unsigned short evL[TT * DE];
  __shared__ __align__(16) unsigned short Plds[8 * 16 * 128];   // 32 KB
  int wave = threadIdx.x >> 5;
  int lane = threadIdx.x & 31, half = lane >> 4, lr = lane & 15;
  int blk = blockIdx.x;                           // B*H*T
  int bh = blk >> 7, t = blk & 127;
  int base, rs;
  if (mode == 0) { base = bh * (TT * TT * DE) + t * (TT * DE); rs = DE; }
  else           { base = bh * (TT * TT * DE) + t * DE;        rs = TT * DE; }
  int m0 = wave * 16;

#if USE_TDM
  if (wave == 0) {
    tdm_load_tile(eqL, eq + base, rs);
    tdm_load_tile(ekL, ek + base, rs);
    tdm_load_tile(evL, ev + base, rs);
    __builtin_amdgcn_s_wait_tensorcnt(0);
  }
#else
  {
    int row = threadIdx.x >> 1, c8 = (threadIdx.x & 1) * 8;
    *(uint4*)(eqL + row * DE + c8) = *(const uint4*)(eq + base + row * rs + c8);
    *(uint4*)(ekL + row * DE + c8) = *(const uint4*)(ek + base + row * rs + c8);
    *(uint4*)(evL + row * DE + c8) = *(const uint4*)(ev + base + row * rs + c8);
  }
#endif
  __syncthreads();

  const uint4 z4 = make_uint4(0u, 0u, 0u, 0u);

  // A = eq rows m0..m0+15 (K = 0..15, zero-padded to 32)
  BF16Vec av;
  av.q[0] = *(const uint4*)(eqL + (m0 + lr) * DE + 8 * half);   // K = 8*half..+7
  av.q[1] = z4;                                                 // K >= 16 -> 0

  v8f accS[8];
  #pragma unroll
  for (int nt = 0; nt < 8; ++nt) {
    BF16Vec bvv;
    const uint4* bp = (const uint4*)(ekL + (nt * 16 + lr) * DE);
    bvv.q[0] = bp[half];                          // K = 8*half..+7
    bvv.q[1] = z4;                                // K >= 16 -> 0
    v8f z = {};
    accS[nt] = wmma_bf16(av.v, bvv.v, z);
  }

  // row softmax: a logical row lives in 16 lanes (one half) x 8 tiles
  #pragma unroll
  for (int r = 0; r < 8; ++r) {
    float mx = -1e30f;
    #pragma unroll
    for (int nt = 0; nt < 8; ++nt) { accS[nt][r] *= 0.25f; mx = fmaxf(mx, accS[nt][r]); }
    for (int msk = 1; msk < 16; msk <<= 1) mx = fmaxf(mx, __shfl_xor(mx, msk, 16));
    float sm = 0.0f;
    #pragma unroll
    for (int nt = 0; nt < 8; ++nt) { float p = __expf(accS[nt][r] - mx); accS[nt][r] = p; sm += p; }
    for (int msk = 1; msk < 16; msk <<= 1) sm += __shfl_xor(sm, msk, 16);
    float inv = 1.0f / sm;
    #pragma unroll
    for (int nt = 0; nt < 8; ++nt) accS[nt][r] *= inv;
  }

  unsigned short* Pw = Plds + wave * 16 * 128;
  #pragma unroll
  for (int nt = 0; nt < 8; ++nt)
    #pragma unroll
    for (int r = 0; r < 8; ++r)
      Pw[(r + 8 * half) * 128 + nt * 16 + lr] = f2bf(accS[nt][r]);
  __syncthreads();

  v8f accO = {};
  #pragma unroll
  for (int kc = 0; kc < 4; ++kc) {
    BF16Vec a2v, b2v;
    const unsigned short* pp = Pw + lr * 128 + kc * 32;
    a2v.q[0] = *(const uint4*)(pp + 8 * half);
    a2v.q[1] = *(const uint4*)(pp + 16 + 8 * half);
    #pragma unroll
    for (int i = 0; i < 16; ++i)
      b2v.u[i] = evL[(kc * 32 + b_kidx(i, half)) * DE + lr];
    accO = wmma_bf16(a2v.v, b2v.v, accO);
  }
  float* dst = eo + base;
  #pragma unroll
  for (int r = 0; r < 8; ++r) {
    int row = m0 + r + 8 * half;
    int idx = row * rs + lr;
    if (mode == 0) dst[idx] = accO[r];
    else           dst[idx] += accO[r];
  }
}

// ---------------------------------------------------------------------------
// K5: eo_lin = eo_r @ Wel + bel.  M=32768, K=128, N=128. A gathered from eo.
// row m = ((b*T + t2)*T + t1); A[m][h*16+de] = eo[b,h,t1,t2,de]
// ---------------------------------------------------------------------------
__global__ void __launch_bounds__(128) eolin_kernel(
    const float* __restrict__ eo, const float* __restrict__ Wel,
    const float* __restrict__ bel, float* __restrict__ eolin)
{
  int gw = blockIdx.x * 4 + (threadIdx.x >> 5);   // 0..16383
  int lane = threadIdx.x & 31, half = lane >> 4, lr = lane & 15;
  int nt = gw & 7, mt = gw >> 3;
  int m0 = mt * 16;
  int row = m0 + lr;
  int t1 = row & 127, t2 = (row >> 7) & 127, b = row >> 14;
  int abase = b * 2097152 + t1 * 2048 + t2 * 16;  // + h*262144 + de
  v8f acc = {};
  #pragma unroll
  for (int kc = 0; kc < 4; ++kc) {
    BF16Vec av, bvv;
    #pragma unroll
    for (int i = 0; i < 16; ++i) {
      int K = kc * 32 + a_kidx(i, half);
      av.u[i] = f2bf(eo[abase + (K >> 4) * 262144 + (K & 15)]);
    }
    #pragma unroll
    for (int i = 0; i < 16; ++i)
      bvv.u[i] = f2bf(Wel[(kc * 32 + b_kidx(i, half)) * EE + nt * 16 + lr]);
    acc = wmma_bf16(av.v, bvv.v, acc);
  }
  int n = nt * 16 + lr;
  float bn = bel[n];
  #pragma unroll
  for (int r = 0; r < 8; ++r) {
    int m = m0 + r + 8 * half;
    eolin[m * EE + n] = acc[r] + bn;
  }
}

// ---------------------------------------------------------------------------
// K6: emb = eo_lin @ Wee + bee.  M=32768, K=128, N=16 (output #2).
// ---------------------------------------------------------------------------
__global__ void __launch_bounds__(128) emb_kernel(
    const float* __restrict__ eolin, const float* __restrict__ Wee,
    const float* __restrict__ bee, float* __restrict__ emb)
{
  int gw = blockIdx.x * 4 + (threadIdx.x >> 5);   // 0..2047 (mt)
  int lane = threadIdx.x & 31, half = lane >> 4, lr = lane & 15;
  int m0 = gw * 16;
  v8f acc = {};
  #pragma unroll
  for (int kc = 0; kc < 4; ++kc) {
    BF16Vec av, bvv;
    load_a_f32(av, eolin + (m0 + lr) * EE + kc * 32, half);
    #pragma unroll
    for (int i = 0; i < 16; ++i)
      bvv.u[i] = f2bf(Wee[(kc * 32 + b_kidx(i, half)) * DE + lr]);
    acc = wmma_bf16(av.v, bvv.v, acc);
  }
  float bn = bee[lr];
  #pragma unroll
  for (int r = 0; r < 8; ++r) {
    int m = m0 + r + 8 * half;
    emb[m * DE + lr] = acc[r] + bn;
  }
}

// ---------------------------------------------------------------------------
// K7: eqk1/eqk2 contractions (K=128 over s or t) + softmax over DE.
// e_out[b,h,t,s,e] = eolin[((b*T+s)*T+t)*E + h*16+e]
// ---------------------------------------------------------------------------
__global__ void __launch_bounds__(256) eqk_kernel(
    const float* __restrict__ sqk, const float* __restrict__ eolin,
    float* __restrict__ a1, float* __restrict__ a2)
{
  int idx = blockIdx.x * 256 + threadIdx.x;       // 0..4095
  int mode = idx >> 11;
  int r = idx & 2047;                             // bh*128 + tt
  int bh = r >> 7, tt = r & 127;
  int b = bh >> 3, h = bh & 7;
  float acc[16];
  #pragma unroll
  for (int e2 = 0; e2 < 16; ++e2) acc[e2] = 0.0f;
  if (mode == 0) {                                 // eqk1: sum over s, fixed t=tt
    for (int s = 0; s < TT; ++s) {
      float q = sqk[bh * TT * TT + s * TT + tt];
      const float* p = eolin + ((b * TT + s) * TT + tt) * EE + h * DE;
      #pragma unroll
      for (int e2 = 0; e2 < 16; ++e2) acc[e2] += q * p[e2];
    }
  } else {                                         // eqk2: sum over t, fixed s=tt
    for (int t_ = 0; t_ < TT; ++t_) {
      float q = sqk[bh * TT * TT + tt * TT + t_];
      const float* p = eolin + ((b * TT + tt) * TT + t_) * EE + h * DE;
      #pragma unroll
      for (int e2 = 0; e2 < 16; ++e2) acc[e2] += q * p[e2];
    }
  }
  const float inv = 0.08838834764831845f;          // 1/sqrt(128)
  float mx = -1e30f;
  #pragma unroll
  for (int e2 = 0; e2 < 16; ++e2) { acc[e2] *= inv; mx = fmaxf(mx, acc[e2]); }
  float sm = 0.0f;
  #pragma unroll
  for (int e2 = 0; e2 < 16; ++e2) { acc[e2] = __expf(acc[e2] - mx); sm += acc[e2]; }
  float nrm = 1.0f / sm;
  float* dst = (mode == 0 ? a1 : a2) + r * DE;
  #pragma unroll
  for (int e2 = 0; e2 < 16; ++e2) dst[e2] = acc[e2] * nrm;
}

// ---------------------------------------------------------------------------
// K8: oe[bh,s,t] = sum_e (a1[bh,t,e] + a2[bh,s,e]) * e_out[t,s,e]
// ---------------------------------------------------------------------------
__global__ void __launch_bounds__(256) oe_kernel(
    const float* __restrict__ a1, const float* __restrict__ a2,
    const float* __restrict__ eolin, float* __restrict__ oe)
{
  int idx = blockIdx.x * 256 + threadIdx.x;       // B*H*T*T
  int t_ = idx & 127, s = (idx >> 7) & 127, bh = idx >> 14;
  int b = bh >> 3, h = bh & 7;
  const float* p  = eolin + ((b * TT + s) * TT + t_) * EE + h * DE;
  const float* p1 = a1 + (bh * TT + t_) * DE;
  const float* p2 = a2 + (bh * TT + s) * DE;
  float acc = 0.0f;
  #pragma unroll
  for (int e2 = 0; e2 < 16; ++e2) acc += (p1[e2] + p2[e2]) * p[e2];
  oe[bh * TT * TT + s * TT + t_] = acc;
}

// ---------------------------------------------------------------------------
// K9: attn[b,s,h*64+d] = sum_t oe[bh,s,t] * v[bh,t,d]   (128x64x128 per bh)
// ---------------------------------------------------------------------------
__global__ void __launch_bounds__(128) attnv_kernel(
    const float* __restrict__ oe, const unsigned short* __restrict__ vb,
    float* __restrict__ attn)
{
  int gw = blockIdx.x * 4 + (threadIdx.x >> 5);   // 0..511
  int lane = threadIdx.x & 31, half = lane >> 4, lr = lane & 15;
  int nt = gw & 3, mt = (gw >> 2) & 7, bh = gw >> 5;
  int b = bh >> 3, h = bh & 7;
  int m0 = mt * 16;
  v8f acc = {};
  #pragma unroll
  for (int kc = 0; kc < 4; ++kc) {
    BF16Vec av, bvv;
    load_a_f32(av, oe + bh * TT * TT + (m0 + lr) * TT + kc * 32, half);
    #pragma unroll
    for (int i = 0; i < 16; ++i)
      bvv.u[i] = vb[bh * TT * DD + (kc * 32 + b_kidx(i, half)) * DD + nt * 16 + lr];
    acc = wmma_bf16(av.v, bvv.v, acc);
  }
  int d = nt * 16 + lr;
  #pragma unroll
  for (int r = 0; r < 8; ++r) {
    int s = m0 + r + 8 * half;
    attn[(b * TT + s) * CC + h * DD + d] = acc[r];
  }
}

// ---------------------------------------------------------------------------
// K10: out = attn @ Wo + bo.  M=256, K=512, N=64 (output #1).
// ---------------------------------------------------------------------------
__global__ void __launch_bounds__(128) out_kernel(
    const float* __restrict__ attn, const float* __restrict__ Wo,
    const float* __restrict__ bo, float* __restrict__ out)
{
  int gw = blockIdx.x * 4 + (threadIdx.x >> 5);   // 0..63
  int lane = threadIdx.x & 31, half = lane >> 4, lr = lane & 15;
  int nt = gw & 3, mt = gw >> 2;
  int m0 = mt * 16;
  v8f acc = {};
  #pragma unroll
  for (int kc = 0; kc < 16; ++kc) {
    BF16Vec av, bvv;
    load_a_f32(av, attn + (m0 + lr) * CC + kc * 32, half);
    #pragma unroll
    for (int i = 0; i < 16; ++i)
      bvv.u[i] = f2bf(Wo[(kc * 32 + b_kidx(i, half)) * DD + nt * 16 + lr]);
    acc = wmma_bf16(av.v, bvv.v, acc);
  }
  int n = nt * 16 + lr;
  float bn = bo[n];
  #pragma unroll
  for (int r = 0; r < 8; ++r) {
    int m = m0 + r + 8 * half;
    out[m * DD + n] = acc[r] + bn;
  }
}

// ---------------------------------------------------------------------------
extern "C" void kernel_launch(void* const* d_in, const int* in_sizes, int n_in,
                              void* d_out, int out_size, void* d_ws, size_t ws_size,
                              hipStream_t stream)
{
  (void)in_sizes; (void)n_in; (void)out_size; (void)ws_size;
  const float* x1  = (const float*)d_in[0];
  const float* e   = (const float*)d_in[1];
  const float* Wq  = (const float*)d_in[2];  const float* bq  = (const float*)d_in[3];
  const float* Wk  = (const float*)d_in[4];  const float* bk  = (const float*)d_in[5];
  const float* Wv  = (const float*)d_in[6];  const float* bv  = (const float*)d_in[7];
  const float* Wo  = (const float*)d_in[8];  const float* bo  = (const float*)d_in[9];
  const float* Weq = (const float*)d_in[10]; const float* beq = (const float*)d_in[11];
  const float* Wev = (const float*)d_in[12]; const float* bev = (const float*)d_in[13];
  const float* Wek = (const float*)d_in[14]; const float* bek = (const float*)d_in[15];
  const float* Wel = (const float*)d_in[16]; const float* bel = (const float*)d_in[17];
  const float* Wee = (const float*)d_in[18]; const float* bee = (const float*)d_in[19];

  float* out = (float*)d_out;               // (B,T,D) = 16384 floats
  float* emb = out + BB * TT * DD;          // (B,T,T,DE) = 524288 floats

  // workspace carve-out (~60 MB)
  char* ws = (char*)d_ws;
  size_t off = 0;
  auto carve = [&](size_t bytes) -> void* {
    void* p = (void*)(ws + off);
    off = (off + bytes + 255) & ~(size_t)255;
    return p;
  };
  unsigned short* qb    = (unsigned short*)carve((size_t)BB * HH * TT * DD * 2);
  unsigned short* kb    = (unsigned short*)carve((size_t)BB * HH * TT * DD * 2);
  unsigned short* vb    = (unsigned short*)carve((size_t)BB * HH * TT * DD * 2);
  float*          sqk   = (float*)carve((size_t)BB * HH * TT * TT * 4);
  unsigned short* eqb   = (unsigned short*)carve((size_t)BB * HH * TT * TT * DE * 2);
  unsigned short* ekb   = (unsigned short*)carve((size_t)BB * HH * TT * TT * DE * 2);
  unsigned short* evb   = (unsigned short*)carve((size_t)BB * HH * TT * TT * DE * 2);
  float*          eo    = (float*)carve((size_t)BB * HH * TT * TT * DE * 4);
  float*          eolin = (float*)carve((size_t)BB * TT * TT * EE * 4);
  float*          a1    = (float*)carve((size_t)BB * HH * TT * DE * 4);
  float*          a2    = (float*)carve((size_t)BB * HH * TT * DE * 4);
  float*          oeb   = (float*)carve((size_t)BB * HH * TT * TT * 4);
  float*          attn  = (float*)carve((size_t)BB * TT * CC * 4);

  qkv_kernel<<<384, 128, 0, stream>>>(x1, Wq, bq, Wk, bk, Wv, bv, qb, kb, vb);
  sqk_kernel<<<256, 128, 0, stream>>>(qb, kb, sqk);
  edge_proj_kernel<<<16384, 256, 0, stream>>>(e, sqk, Weq, beq, Wev, bev, Wek, bek,
                                              eqb, evb, ekb);
  edge_attn_kernel<<<BB * HH * TT, 256, 0, stream>>>(eqb, ekb, evb, eo, 0);
  edge_attn_kernel<<<BB * HH * TT, 256, 0, stream>>>(eqb, ekb, evb, eo, 1);
  eolin_kernel<<<4096, 128, 0, stream>>>(eo, Wel, bel, eolin);
  emb_kernel<<<512, 128, 0, stream>>>(eolin, Wee, bee, emb);
  eqk_kernel<<<16, 256, 0, stream>>>(sqk, eolin, a1, a2);
  oe_kernel<<<1024, 256, 0, stream>>>(a1, a2, eolin, oeb);
  attnv_kernel<<<128, 128, 0, stream>>>(oeb, vb, attn);
  out_kernel<<<16, 128, 0, stream>>>(attn, Wo, bo, out);
}
